// GraphVariationalAutoEncoder_79104707657805
// MI455X (gfx1250) — matrix operand order repaired and checked by reading, not compiled, not run on hardware
//
#include <hip/hip_runtime.h>
#include <hip/hip_bf16.h>

#define HID 128

typedef __attribute__((ext_vector_type(16))) _Float16 v16h;
typedef __attribute__((ext_vector_type(8)))  _Float16 v8h;
typedef __attribute__((ext_vector_type(8)))  float    v8f;
typedef __attribute__((ext_vector_type(4)))  float    v4f;

union V16H { v16h v; v8h h[2]; };

// ---------------- degree / normalization ----------------

__global__ void k_fill1(float* p, int n) {
    int i = blockIdx.x * blockDim.x + threadIdx.x;
    if (i < n) p[i] = 1.0f;   // self-loop contributes 1 to every degree
}

__global__ void k_deg(const int* __restrict__ dst, float* deg, int E) {
    int e = blockIdx.x * blockDim.x + threadIdx.x;
    if (e < E) atomicAdd(&deg[dst[e]], 1.0f);
}

__global__ void k_rsqrt(float* p, int n) {
    int i = blockIdx.x * blockDim.x + threadIdx.x;
    if (i < n) p[i] = rsqrtf(p[i]);   // deg >= 1 always (self-loop)
}

// ---------------- WMMA GEMM: Y[nRows,128] = X[nRows,128] @ W[128,128] (+bias) ----------------
// 8 waves/block, each wave computes a 16x128 tile via 32x v_wmma_f32_16x16x32_f16.
// W^T staged in LDS as f16 with padded pitch (136 halves) for conflict-free B loads.
// All A-fragment global loads for the whole K loop are issued up front so they
// pipeline (one wait amortized over 16 b128 loads) instead of serializing.

__global__ __launch_bounds__(256) void k_gemm(const float* __restrict__ X,
                                              const float* __restrict__ W,
                                              const float* __restrict__ bias,
                                              float* __restrict__ Y, int nRows) {
    __shared__ _Float16 wT[128 * 136];

    const int t = threadIdx.x;
    // stage W (row-major [k][n], fp32) transposed into LDS as f16: wT[n*136 + k]
    for (int idx = t; idx < 128 * 128; idx += 256) {
        int k = idx >> 7, n = idx & 127;
        wT[n * 136 + k] = (_Float16)W[idx];
    }
    __syncthreads();

    const int lane  = t & 31;
    const int wave  = t >> 5;                       // 0..7
    const int tileRow = blockIdx.x * 128 + wave * 16;
    const int mLane = lane & 15;                    // row (A) / col (B,C) within 16-tile
    const int hi    = lane >> 4;                    // half-wave select

    // A row for this lane (clamped for tail; keeps EXEC all-1s for WMMA)
    int rowA = tileRow + mLane;
    if (rowA > nRows - 1) rowA = nRows - 1;
    const float* xrow = X + (size_t)rowA * HID;
    const int offA = hi * 8;                        // lanes16-31 start at K=8 within tile

    // ---- prefetch: all 64 floats this lane contributes across the 4 K-steps ----
    v4f xr[16];
    #pragma unroll
    for (int kt = 0; kt < 4; ++kt) {
        const float* p0 = xrow + kt * 32 + offA;
        xr[kt * 4 + 0] = *(const v4f*)(p0);
        xr[kt * 4 + 1] = *(const v4f*)(p0 + 4);
        xr[kt * 4 + 2] = *(const v4f*)(p0 + 16);
        xr[kt * 4 + 3] = *(const v4f*)(p0 + 20);
    }

    v8f acc[8] = {};

    #pragma unroll
    for (int kt = 0; kt < 4; ++kt) {
        // --- A fragment: 16-bit A 16x32 layout ---
        // lane<16 : elems 0..7 = K0..7,  elems 8..15 = K16..23
        // lane>=16: elems 0..7 = K8..15, elems 8..15 = K24..31
        V16H A;
        #pragma unroll
        for (int i = 0; i < 4; ++i) {
            A.v[i]      = (_Float16)xr[kt * 4 + 0][i];
            A.v[4 + i]  = (_Float16)xr[kt * 4 + 1][i];
            A.v[8 + i]  = (_Float16)xr[kt * 4 + 2][i];
            A.v[12 + i] = (_Float16)xr[kt * 4 + 3][i];
        }
        const int kb = kt * 32 + hi * 8;            // B: lanes0-15 K0..7,K16..23, lanes16-31 K8..15,K24..31
        #pragma unroll
        for (int n = 0; n < 8; ++n) {
            const _Float16* wp = &wT[(n * 16 + mLane) * 136 + kb];
            V16H B;
            B.h[0] = *(const v8h*)(wp);
            B.h[1] = *(const v8h*)(wp + 16);
            acc[n] = __builtin_amdgcn_wmma_f32_16x16x32_f16(
                false, A.v, false, B.v, (short)0, acc[n], false, false);
        }
    }

    // C/D layout: VGPR v -> M = v + 8*hi, N = mLane
    #pragma unroll
    for (int n = 0; n < 8; ++n) {
        const int col = n * 16 + mLane;
        float badd = bias ? bias[col] : 0.0f;
        #pragma unroll
        for (int v = 0; v < 8; ++v) {
            int r = tileRow + v + hi * 8;
            if (r < nRows) Y[(size_t)r * HID + col] = acc[n][v] + badd;
        }
    }
}

// ---------------- aggregation ----------------

// self-loop term: agg[i,f] = h[i,f] * dis[i]^2
__global__ void k_self(const float* __restrict__ h, const float* __restrict__ dis,
                       float* __restrict__ agg, int n128) {
    int i = blockIdx.x * blockDim.x + threadIdx.x;
    if (i < n128) {
        float d = dis[i >> 7];
        agg[i] = h[i] * d * d;
    }
}

// edge scatter: 32 lanes per edge, float4 per lane (128 feats), atomics hit L2 (data L2-resident)
__global__ void k_scatter(const int* __restrict__ src, const int* __restrict__ dst,
                          const float* __restrict__ dis, const float* __restrict__ h,
                          float* agg, int E) {
    int t = blockIdx.x * blockDim.x + threadIdx.x;
    int e = t >> 5;
    if (e >= E) return;
    int q = t & 31;
    int s = src[e], d = dst[e];
    float norm = dis[s] * dis[d];
    v4f v = *(const v4f*)(h + (size_t)s * HID + q * 4);
    float* ap = agg + (size_t)d * HID + q * 4;
    atomicAdd(ap + 0, v[0] * norm);
    atomicAdd(ap + 1, v[1] * norm);
    atomicAdd(ap + 2, v[2] * norm);
    atomicAdd(ap + 3, v[3] * norm);
}

__global__ void k_bias_relu(float* agg, const float* __restrict__ b, int n128) {
    int i = blockIdx.x * blockDim.x + threadIdx.x;
    if (i < n128) {
        float v = agg[i] + b[i & 127];
        agg[i] = v > 0.0f ? v : 0.0f;
    }
}

// ---------------- launch ----------------

extern "C" void kernel_launch(void* const* d_in, const int* in_sizes, int n_in,
                              void* d_out, int out_size, void* d_ws, size_t ws_size,
                              hipStream_t stream) {
    const float* x  = (const float*)d_in[0];
    const int*   ei = (const int*)d_in[1];
    const float* W0 = (const float*)d_in[2];
    const float* b0 = (const float*)d_in[3];
    const float* W1 = (const float*)d_in[4];
    const float* b1 = (const float*)d_in[5];
    const float* W2 = (const float*)d_in[6];
    const float* b2 = (const float*)d_in[7];
    const float* Wd = (const float*)d_in[8];
    const float* bd = (const float*)d_in[9];

    const int N = in_sizes[0] / HID;
    const int E = in_sizes[1] / 2;
    const int* srcp = ei;
    const int* dstp = ei + E;

    char* ws = (char*)d_ws;
    float* dis = (float*)ws;                                        // N floats
    size_t offH = ((size_t)N * sizeof(float) + 4095) & ~(size_t)4095;
    float* h   = (float*)(ws + offH);                               // N*128 floats
    float* agg = (float*)(ws + offH + (size_t)N * HID * sizeof(float)); // N*128 floats

    const int n128 = N * HID;

    // degree / normalization
    k_fill1<<<(N + 255) / 256, 256, 0, stream>>>(dis, N);
    k_deg<<<(E + 255) / 256, 256, 0, stream>>>(dstp, dis, E);
    k_rsqrt<<<(N + 255) / 256, 256, 0, stream>>>(dis, N);

    dim3 gGemm((N + 127) / 128);
    dim3 gScat(((long long)E * 32 + 255) / 256);

    const float* in = x;
    const float* Ws[3] = {W0, W1, W2};
    const float* bs[3] = {b0, b1, b2};
    for (int l = 0; l < 3; ++l) {
        k_gemm<<<gGemm, 256, 0, stream>>>(in, Ws[l], nullptr, h, N);
        k_self<<<(n128 + 255) / 256, 256, 0, stream>>>(h, dis, agg, n128);
        k_scatter<<<gScat, 256, 0, stream>>>(srcp, dstp, dis, h, agg, E);
        k_bias_relu<<<(n128 + 255) / 256, 256, 0, stream>>>(agg, bs[l], n128);
        in = agg;
    }
    // dense head: d_out = agg @ Wd + bd
    k_gemm<<<gGemm, 256, 0, stream>>>(in, Wd, bd, (float*)d_out, N);
}
